// TopK_Sparse_Attention_9363028706190
// MI455X (gfx1250) — matrix operand order repaired and checked by reading, not compiled, not run on hardware
//
#include <hip/hip_runtime.h>
#include <hip/hip_bf16.h>
#include <stdint.h>

// ---------------------------------------------------------------------------
// TopK sparse channel attention, MI455X (gfx1250, wave32, WMMA bf16).
// Memory-bound workload (~90 GFLOP vs ~0.8 GB traffic @ 23.3 TB/s), so all
// large intermediates are bf16; all GEMMs use v_wmma_f32_16x16x32_bf16 with
// fp32 accumulation. The four top-k softmaxes are fused into one combined
// P matrix so P@V is a single GEMM.
// Workspace requirement: ~353 MB.
// ---------------------------------------------------------------------------

#define NTOK 16384   // 128*128 tokens
#define NB   4
#define CDIM 384
#define QKVC 1152
#define NH   8
#define CH   48

typedef __attribute__((ext_vector_type(16))) __bf16 v16bf;
typedef __attribute__((ext_vector_type(8)))  float  v8f;

union FragAB { uint4 q[2]; v16bf v; };

__device__ __forceinline__ unsigned short f2bf(float f) {
  unsigned u = __float_as_uint(f);
  u += 0x7fffu + ((u >> 16) & 1u);           // round-to-nearest-even
  return (unsigned short)(u >> 16);
}
__device__ __forceinline__ float bf2f(unsigned short s) {
  return __uint_as_float(((unsigned)s) << 16);
}
__device__ __forceinline__ float bfl(unsigned v) { return __uint_as_float(v << 16); }
__device__ __forceinline__ float bfh(unsigned v) { return __uint_as_float(v & 0xffff0000u); }

// ---------------------------------------------------------------------------
// Generic GEMM:  Y[b][Mtot][NTOK] = Wf32[Mtot][Kdim] * X[b][Kdim][NTOK]
// Block tile 128(M) x 64(N), K-step 32, 8 waves (each wave: 16 x 64 strip).
// ---------------------------------------------------------------------------
template <bool IN_BF16, bool OUT_BF16>
__global__ __launch_bounds__(256)
void gemm_wmma(const float* __restrict__ Wmat,
               const void*  __restrict__ Xv,
               void*        __restrict__ Yv,
               int Mtot, int Kdim)
{
  __shared__ __align__(16) unsigned short As[128 * 32]; // [m][k] bf16
  __shared__ __align__(16) unsigned short Bs[64 * 32];  // [n][k] bf16 (transposed)

  const int tid  = threadIdx.x;
  const int w    = tid >> 5;
  const int lane = tid & 31;
  const int lm   = lane & 15;
  const int hi   = lane >> 4;
  const int kbA  = hi ? 8 : 0;   // A frag: lane<16 -> K{0..7,16..23}
  const int kbB  = hi ? 8 : 0;   // B frag: lane<16 -> K{0..7,16..23}

  const int nb    = blockIdx.x * 64;
  const int mbase = blockIdx.y * 128;
  const int bz    = blockIdx.z;

  v8f acc[4] = {};

  const int ksteps = Kdim >> 5;
  for (int kt = 0; kt < ksteps; ++kt) {
    const int k0 = kt << 5;
    __syncthreads();
    // ---- stage A: 128x32 weights, f32 -> bf16
    for (int ci = tid; ci < 1024; ci += 256) {
      const int row = ci >> 3, c4 = (ci & 7) << 2;
      const float4 f = *reinterpret_cast<const float4*>(
          Wmat + (size_t)(mbase + row) * Kdim + k0 + c4);
      unsigned short* d = As + row * 32 + c4;
      d[0] = f2bf(f.x); d[1] = f2bf(f.y); d[2] = f2bf(f.z); d[3] = f2bf(f.w);
    }
    // ---- stage B: 32x64 of X, transposed to [n][k]
    if (IN_BF16) {
      const unsigned short* X = (const unsigned short*)Xv;
      for (int ci = tid; ci < 256; ci += 256) {
        const int kr = ci >> 3, n8 = (ci & 7) << 3;
        const uint4 u = *reinterpret_cast<const uint4*>(
            X + ((size_t)bz * Kdim + k0 + kr) * NTOK + nb + n8);
        const unsigned uu[4] = {u.x, u.y, u.z, u.w};
        #pragma unroll
        for (int j = 0; j < 4; ++j) {
          Bs[(n8 + 2 * j + 0) * 32 + kr] = (unsigned short)(uu[j] & 0xffffu);
          Bs[(n8 + 2 * j + 1) * 32 + kr] = (unsigned short)(uu[j] >> 16);
        }
      }
    } else {
      const float* X = (const float*)Xv;
      for (int ci = tid; ci < 512; ci += 256) {
        const int kr = ci >> 4, n4 = (ci & 15) << 2;
        const float4 f = *reinterpret_cast<const float4*>(
            X + ((size_t)bz * Kdim + k0 + kr) * NTOK + nb + n4);
        Bs[(n4 + 0) * 32 + kr] = f2bf(f.x);
        Bs[(n4 + 1) * 32 + kr] = f2bf(f.y);
        Bs[(n4 + 2) * 32 + kr] = f2bf(f.z);
        Bs[(n4 + 3) * 32 + kr] = f2bf(f.w);
      }
    }
    if (kt + 1 < ksteps) {   // advisory prefetch of next weight slab
      __builtin_prefetch(Wmat + (size_t)(mbase + (tid & 127)) * Kdim + k0 + 32, 0, 1);
    }
    __syncthreads();

    FragAB a;
    const unsigned short* ar = As + (w * 16 + lm) * 32;
    a.q[0] = *reinterpret_cast<const uint4*>(ar + kbA);
    a.q[1] = *reinterpret_cast<const uint4*>(ar + kbA + 16);
    #pragma unroll
    for (int s = 0; s < 4; ++s) {
      FragAB bfr;
      const unsigned short* br = Bs + (s * 16 + lm) * 32 + kbB;
      bfr.q[0] = *reinterpret_cast<const uint4*>(br);
      bfr.q[1] = *reinterpret_cast<const uint4*>(br + 16);
      acc[s] = __builtin_amdgcn_wmma_f32_16x16x32_bf16(
          false, a.v, false, bfr.v, (short)0, acc[s], false, false);
    }
  }

  #pragma unroll
  for (int s = 0; s < 4; ++s)
    #pragma unroll
    for (int r = 0; r < 8; ++r) {
      const int mloc = (hi ? 8 : 0) + r;  // C/D layout: VGPR r -> M=r (+8 upper half)
      const size_t idx =
          ((size_t)bz * Mtot + mbase + w * 16 + mloc) * NTOK + nb + s * 16 + lm;
      if (OUT_BF16) ((unsigned short*)Yv)[idx] = f2bf(acc[s][r]);
      else          ((float*)Yv)[idx]          = acc[s][r];
    }
}

// ---------------------------------------------------------------------------
// Depthwise 3x3, SAME padding, bf16 in/out, fp32 math.
// ---------------------------------------------------------------------------
__global__ __launch_bounds__(256)
void dwconv3x3(const unsigned short* __restrict__ in, const float* __restrict__ wts,
               unsigned short* __restrict__ out)
{
  const int ch = blockIdx.y, b = blockIdx.z;
  const int t = threadIdx.x;
  const int x = t & 127, y = blockIdx.x * 2 + (t >> 7);
  const unsigned short* p = in + ((size_t)b * QKVC + ch) * NTOK;
  const float* wc = wts + ch * 9;
  float s = 0.f;
  #pragma unroll
  for (int dy = -1; dy <= 1; ++dy) {
    const int yy = y + dy; if (yy < 0 || yy > 127) continue;
    #pragma unroll
    for (int dx = -1; dx <= 1; ++dx) {
      const int xx = x + dx; if (xx < 0 || xx > 127) continue;
      s += bf2f(p[yy * 128 + xx]) * wc[(dy + 1) * 3 + (dx + 1)];
    }
  }
  out[((size_t)b * QKVC + ch) * NTOK + y * 128 + x] = f2bf(s);
}

// ---------------------------------------------------------------------------
// 1/L2-norm per q/k row (rows 0..767 of post-dw qkv).
// ---------------------------------------------------------------------------
__global__ __launch_bounds__(256)
void rownorm(const unsigned short* __restrict__ qkv, float* __restrict__ invn)
{
  __shared__ float red[256];
  const int o = blockIdx.x, b = blockIdx.y, tid = threadIdx.x;
  const unsigned short* row = qkv + ((size_t)b * QKVC + o) * NTOK;
  float s = 0.f;
  for (int j = tid * 8; j < NTOK; j += 256 * 8) {
    const uint4 u = *reinterpret_cast<const uint4*>(row + j);
    const unsigned uu[4] = {u.x, u.y, u.z, u.w};
    #pragma unroll
    for (int q = 0; q < 4; ++q) {
      const float lo = bfl(uu[q]), hv = bfh(uu[q]);
      s += lo * lo + hv * hv;
    }
  }
  red[tid] = s; __syncthreads();
  for (int st = 128; st > 0; st >>= 1) {
    if (tid < st) red[tid] += red[tid + st];
    __syncthreads();
  }
  if (tid == 0) invn[b * 768 + o] = rsqrtf(red[0]);
}

// ---------------------------------------------------------------------------
// attn = Qn Kn^T * temp (48x48 per (b,h)), then fused 4-way top-k softmax:
// Psum = sum_i a_i * softmax(mask_{k_i}(attn)).  K=16384 split over 8 waves,
// partials reduced via LDS ds_add_f32.  Psum stored bf16 [48][64] (zero pad).
// ---------------------------------------------------------------------------
__global__ __launch_bounds__(256)
void attn_topk(const unsigned short* __restrict__ qkv,
               const float* __restrict__ invn,
               const float* __restrict__ temp,
               const float* __restrict__ a1, const float* __restrict__ a2,
               const float* __restrict__ a3, const float* __restrict__ a4,
               unsigned short* __restrict__ Psum)
{
  __shared__ float attnS[48 * 48];
  __shared__ float work[48 * 48];

  const int h = blockIdx.x, b = blockIdx.y;
  const int tid = threadIdx.x, w = tid >> 5, lane = tid & 31;
  const int lm = lane & 15, hi = lane >> 4;
  const int kbA = hi ? 8 : 0, kbB = hi ? 8 : 0;

  for (int i = tid; i < 48 * 48; i += 256) attnS[i] = 0.f;
  __syncthreads();

  const unsigned short* qb = qkv + ((size_t)b * QKVC + h * CH) * NTOK;
  const unsigned short* kp = qkv + ((size_t)b * QKVC + CDIM + h * CH) * NTOK;

  v8f acc[9] = {};
  const int base = w * (NTOK / 8);
  for (int it = 0; it < (NTOK / 8) / 32; ++it) {
    const int n0 = base + it * 32;
    FragAB af[3], bfm[3];
    #pragma unroll
    for (int t = 0; t < 3; ++t) {
      const unsigned short* ap = qb + (size_t)(t * 16 + lm) * NTOK + n0 + kbA;
      af[t].q[0] = *reinterpret_cast<const uint4*>(ap);
      af[t].q[1] = *reinterpret_cast<const uint4*>(ap + 16);
      const unsigned short* bp = kp + (size_t)(t * 16 + lm) * NTOK + n0 + kbB;
      bfm[t].q[0] = *reinterpret_cast<const uint4*>(bp);
      bfm[t].q[1] = *reinterpret_cast<const uint4*>(bp + 16);
    }
    #pragma unroll
    for (int mt = 0; mt < 3; ++mt)
      #pragma unroll
      for (int nt = 0; nt < 3; ++nt)
        acc[mt * 3 + nt] = __builtin_amdgcn_wmma_f32_16x16x32_bf16(
            false, af[mt].v, false, bfm[nt].v, (short)0, acc[mt * 3 + nt], false, false);
  }
  #pragma unroll
  for (int mt = 0; mt < 3; ++mt)
    for (int nt = 0; nt < 3; ++nt)
      #pragma unroll
      for (int r = 0; r < 8; ++r) {
        const int i = mt * 16 + (hi ? 8 : 0) + r;
        const int j = nt * 16 + lm;
        atomicAdd(&attnS[i * 48 + j], acc[mt * 3 + nt][r]);   // ds_add_f32
      }
  __syncthreads();

  const float tmp = temp[h];
  for (int idx = tid; idx < 48 * 48; idx += 256) {
    const int i = idx / 48, j = idx % 48;
    attnS[idx] *= invn[b * 768 + h * CH + i] * invn[b * 768 + CDIM + h * CH + j] * tmp;
  }
  __syncthreads();

  if (tid < 48) {
    const int i = tid;
    float* rowS = attnS + i * 48;
    float* rowW = work + i * 48;
    for (int j = 0; j < 48; ++j) rowW[j] = rowS[j];
    const int kw[4] = {24, 32, 36, 38};   // C/2, 2C/3, 3C/4, 4C/5 with C=48
    float thr[4]; float mx = -3.0e38f; int ci = 0;
    for (int t = 0; t < 38; ++t) {        // iterative selection up to 38th largest
      float m = -3.0e38f; int mj = 0;
      for (int j = 0; j < 48; ++j) { const float v = rowW[j]; if (v > m) { m = v; mj = j; } }
      rowW[mj] = -3.0e38f;
      if (t == 0) mx = m;
      if (ci < 4 && (t + 1) == kw[ci]) thr[ci++] = m;
    }
    float den[4] = {0, 0, 0, 0};
    for (int j = 0; j < 48; ++j) {
      const float v = rowS[j];
      const float e = __expf(v - mx);
      #pragma unroll
      for (int c = 0; c < 4; ++c) if (v >= thr[c]) den[c] += e;
    }
    const float aw[4] = {a1[0], a2[0], a3[0], a4[0]};
    unsigned short* pr = Psum + ((size_t)(b * NH + h) * 48 + i) * 64;
    for (int j = 0; j < 48; ++j) {
      const float v = rowS[j];
      const float e = __expf(v - mx);
      float p = 0.f;
      #pragma unroll
      for (int c = 0; c < 4; ++c) if (v >= thr[c]) p += aw[c] * e / den[c];
      pr[j] = f2bf(p);
    }
    for (int j = 48; j < 64; ++j) pr[j] = 0;
  }
}

// ---------------------------------------------------------------------------
// out = Psum @ V  per (b,h):  [48x64pad] @ [64pad x NTOK] -> bf16 [48 x NTOK].
// 6 waves: 3 M-tiles x 2 N-groups of 64; V transposed through LDS.
// ---------------------------------------------------------------------------
__global__ __launch_bounds__(192)
void pv_gemm(const unsigned short* __restrict__ Psum,
             const unsigned short* __restrict__ qkv,
             unsigned short* __restrict__ outb)
{
  __shared__ __align__(16) unsigned short Vt[128 * 64];  // [n][c] bf16
  const int tid = threadIdx.x, w = tid >> 5, lane = tid & 31;
  const int lm = lane & 15, hi = lane >> 4;
  const int kbA = hi ? 8 : 0, kbB = hi ? 8 : 0;
  const int nb = blockIdx.x * 128, h = blockIdx.y, b = blockIdx.z;
  const int wm = w % 3, wn = w / 3;

  const unsigned short* vb = qkv + ((size_t)b * QKVC + 2 * CDIM + h * CH) * NTOK;
  for (int ci = tid; ci < 768; ci += 192) {
    const int c = ci >> 4, n8 = (ci & 15) << 3;
    const uint4 u = *reinterpret_cast<const uint4*>(vb + (size_t)c * NTOK + nb + n8);
    const unsigned uu[4] = {u.x, u.y, u.z, u.w};
    #pragma unroll
    for (int j = 0; j < 4; ++j) {
      Vt[(n8 + 2 * j + 0) * 64 + c] = (unsigned short)(uu[j] & 0xffffu);
      Vt[(n8 + 2 * j + 1) * 64 + c] = (unsigned short)(uu[j] >> 16);
    }
  }
  for (int idx = tid; idx < 128 * 16; idx += 192) {  // zero the K-pad (avoid NaN*0)
    Vt[(idx >> 4) * 64 + 48 + (idx & 15)] = 0;
  }
  __syncthreads();

  const unsigned short* pb = Psum + (size_t)(b * NH + h) * 48 * 64;
  v8f acc[4] = {};
  #pragma unroll
  for (int kt = 0; kt < 2; ++kt) {
    FragAB a;
    const unsigned short* ap = pb + (size_t)(wm * 16 + lm) * 64 + kt * 32 + kbA;
    a.q[0] = *reinterpret_cast<const uint4*>(ap);
    a.q[1] = *reinterpret_cast<const uint4*>(ap + 16);
    #pragma unroll
    for (int s = 0; s < 4; ++s) {
      FragAB bfr;
      const unsigned short* bp = Vt + (wn * 64 + s * 16 + lm) * 64 + kt * 32 + kbB;
      bfr.q[0] = *reinterpret_cast<const uint4*>(bp);
      bfr.q[1] = *reinterpret_cast<const uint4*>(bp + 16);
      acc[s] = __builtin_amdgcn_wmma_f32_16x16x32_bf16(
          false, a.v, false, bfr.v, (short)0, acc[s], false, false);
    }
  }
  #pragma unroll
  for (int s = 0; s < 4; ++s)
    #pragma unroll
    for (int r = 0; r < 8; ++r) {
      const int mloc = (hi ? 8 : 0) + r;
      const int row = h * CH + wm * 16 + mloc;
      outb[((size_t)b * CDIM + row) * NTOK + nb + wn * 64 + s * 16 + lm] = f2bf(acc[s][r]);
    }
}

// ---------------------------------------------------------------------------
extern "C" void kernel_launch(void* const* d_in, const int* in_sizes, int n_in,
                              void* d_out, int out_size, void* d_ws, size_t ws_size,
                              hipStream_t stream)
{
  (void)in_sizes; (void)n_in; (void)out_size; (void)ws_size;
  const float* x      = (const float*)d_in[0];
  const float* qkv_w  = (const float*)d_in[1];
  const float* dw_w   = (const float*)d_in[2];
  const float* proj_w = (const float*)d_in[3];
  const float* temp   = (const float*)d_in[4];
  const float* a1 = (const float*)d_in[5];
  const float* a2 = (const float*)d_in[6];
  const float* a3 = (const float*)d_in[7];
  const float* a4 = (const float*)d_in[8];

  char* ws = (char*)d_ws;
  const size_t qkvBytes = (size_t)NB * QKVC * NTOK * sizeof(unsigned short); // 151 MB
  unsigned short* qkv_lin = (unsigned short*)ws; ws += qkvBytes;
  unsigned short* qkv_dw  = (unsigned short*)ws; ws += qkvBytes;
  float*          invn    = (float*)ws;
  ws += (((size_t)NB * 768 * sizeof(float)) + 255) & ~(size_t)255;
  unsigned short* Ps      = (unsigned short*)ws;
  ws += (((size_t)NB * NH * 48 * 64 * sizeof(unsigned short)) + 255) & ~(size_t)255;
  unsigned short* attn_o  = (unsigned short*)ws;                            // 50 MB

  // 1) QKV 1x1-conv GEMM: [1152,384] x [384,16384] per batch
  gemm_wmma<false, true><<<dim3(256, 9, NB), 256, 0, stream>>>(qkv_w, x, qkv_lin, QKVC, CDIM);
  // 2) depthwise 3x3
  dwconv3x3<<<dim3(64, QKVC, NB), 256, 0, stream>>>(qkv_lin, dw_w, qkv_dw);
  // 3) inverse L2 norms of q,k rows
  rownorm<<<dim3(2 * CDIM, NB), 256, 0, stream>>>(qkv_dw, invn);
  // 4) 48x48 attention + fused 4-way top-k softmax -> Psum
  attn_topk<<<dim3(NH, NB), 256, 0, stream>>>(qkv_dw, invn, temp, a1, a2, a3, a4, Ps);
  // 5) Psum @ V
  pv_gemm<<<dim3(NTOK / 128, NH, NB), 192, 0, stream>>>(Ps, qkv_dw, attn_o);
  // 6) projection GEMM -> fp32 output
  gemm_wmma<true, false><<<dim3(256, 3, NB), 256, 0, stream>>>(proj_w, attn_o, d_out, CDIM, CDIM);
}
